// BBoxDetector_22462678958362
// MI455X (gfx1250) — compile-verified, matrix operand order checked by reading, not run on hardware
//
#include <hip/hip_runtime.h>
#include <hip/hip_bf16.h>

// ---------------------------------------------------------------------------
// Types for WMMA fragments (gfx1250, wave32)
// ---------------------------------------------------------------------------
typedef __attribute__((ext_vector_type(16))) __bf16        v16bf;
typedef __attribute__((ext_vector_type(8)))  float         v8f;
typedef __attribute__((ext_vector_type(4)))  unsigned int  u32x4;

union Frag16 {           // 16 bf16 values = 8 VGPRs, filled via two 16B loads
    u32x4 q[2];
    v16bf v;
};

__device__ __forceinline__ unsigned short f2bf(float f) {
    unsigned u = __float_as_uint(f);
    unsigned r = u + 0x7FFFu + ((u >> 16) & 1u);   // round-to-nearest-even
    return (unsigned short)(r >> 16);
}
__device__ __forceinline__ float bf2f(unsigned short s) {
    return __uint_as_float(((unsigned)s) << 16);
}

#define BD 16
#define HH 56
#define WW 56
#define HWPIX (HH * WW)          // 3136
#define NPIX (BD * HWPIX)        // 50176
#define NANCH 9
#define NBOX (HWPIX * NANCH)     // 28224 per image
#define MAXDET 100
#define NEGF (-1e30f)
#define PADELEMS 512             // zero page: >= max Cin bf16 elements

// ---------------------------------------------------------------------------
// Zero the OOB pad page (re-done every launch; ws may be poisoned)
// ---------------------------------------------------------------------------
__global__ __launch_bounds__(256) void zero_pad(unsigned short* __restrict__ p)
{
    const int i = threadIdx.x;
    p[i * 2 + 0] = 0;
    p[i * 2 + 1] = 0;
}

// ---------------------------------------------------------------------------
// Fold BN into conv weights, repack OIHW f32 -> [tap][cout][cin] bf16
// ---------------------------------------------------------------------------
__global__ __launch_bounds__(256) void fold_weights(
    const float* __restrict__ w, const float* __restrict__ b,
    const float* __restrict__ g, const float* __restrict__ be,
    const float* __restrict__ m, const float* __restrict__ v,
    unsigned short* __restrict__ Wt, float* __restrict__ bias,
    int Cin, int Cout)
{
    int idx = blockIdx.x * 256 + threadIdx.x;          // over Cout*Cin*9
    int total = Cout * Cin * 9;
    if (idx >= total) return;
    int t = idx;
    int kx = t % 3; t /= 3;
    int ky = t % 3; t /= 3;
    int ci = t % Cin;
    int co = t / Cin;
    float inv = g[co] * rsqrtf(v[co] + 1e-5f);
    float wv = w[((size_t)(co * Cin + ci) * 3 + ky) * 3 + kx] * inv;
    Wt[((size_t)((ky * 3 + kx) * Cout + co)) * Cin + ci] = f2bf(wv);
    if (ci == 0 && ky == 0 && kx == 0)
        bias[co] = b[co] * inv + be[co] - m[co] * inv;
}

// ---------------------------------------------------------------------------
// NCHW f32 -> NHWC bf16 repack (makes K contiguous for implicit GEMM)
// ---------------------------------------------------------------------------
__global__ __launch_bounds__(256) void nchw_to_nhwc_bf16(
    const float* __restrict__ X, unsigned short* __restrict__ Y, int C)
{
    size_t idx = (size_t)blockIdx.x * 256 + threadIdx.x;   // output (NHWC) index
    size_t total = (size_t)NPIX * C;
    if (idx >= total) return;
    int c = (int)(idx % C);
    size_t p = idx / C;                // pixel index b*HW + hw
    size_t b = p / HWPIX;
    size_t hw = p - b * HWPIX;
    Y[idx] = f2bf(X[(b * C + c) * HWPIX + hw]);
}

// ---------------------------------------------------------------------------
// Implicit-GEMM 3x3 conv, pad 1, folded BN + ReLU, bf16 in/out, f32 accum.
//   X  : NHWC bf16 [B,H,W,Cin]
//   Wt : [3][3][Cout][Cin] bf16 (BN folded)
// Each wave computes a 32(M) x 64(N) output tile: 2 A fragments x 4 N-tiles
// = 8 WMMA accumulators; every B fragment is reused twice (1.5 loads/WMMA).
// Out-of-bounds taps read a zeroed pad page: inner loop is branch-free.
// ---------------------------------------------------------------------------
__global__ __launch_bounds__(256) void conv3x3_wmma(
    const unsigned short* __restrict__ X,
    const unsigned short* __restrict__ Wt,
    const float* __restrict__ bias,
    const unsigned short* __restrict__ zpad,   // >= Cin zeros (bf16)
    unsigned short* __restrict__ Y,
    int Cin, int Cout)
{
    const int lane = threadIdx.x & 31;
    const int wave = blockIdx.x * 8 + (threadIdx.x >> 5);
    const int nblocks = Cout >> 6;                 // 64-channel N blocks
    const int mblk   = wave / nblocks;             // 32-row M block
    const int nblock = wave - mblk * nblocks;
    const int m0 = mblk * 32;
    const int coutBase = nblock * 64;

    const int half = lane >> 4;                    // K-half select (layout)
    const int mrow = lane & 15;                    // A-matrix row M
    const int nidx = lane & 15;                    // B/C/D column N

    // per-lane pixels for the two A fragments (rows m0..m0+15, m0+16..m0+31)
    const int pix0 = m0 + mrow;
    const int pix1 = m0 + 16 + mrow;
    const int b0 = pix0 / HWPIX;
    const int hw0 = pix0 - b0 * HWPIX;
    const int h0 = hw0 / WW;
    const int w0 = hw0 - h0 * WW;
    const int b1 = pix1 / HWPIX;
    const int hw1 = pix1 - b1 * HWPIX;
    const int h1 = hw1 / WW;
    const int w1 = hw1 - h1 * WW;

    v8f acc[2][4] = {{v8f{}, v8f{}, v8f{}, v8f{}},
                     {v8f{}, v8f{}, v8f{}, v8f{}}};

    const int koff0 = half * 8;                    // first K run offset
    const int koff1 = 16 + half * 8;               // second K run offset

    for (int ky = 0; ky < 3; ++ky) {
        const int hh0 = h0 + ky - 1;
        const int hh1 = h1 + ky - 1;
        const bool hok0 = (unsigned)hh0 < (unsigned)HH;
        const bool hok1 = (unsigned)hh1 < (unsigned)HH;
        for (int kx = 0; kx < 3; ++kx) {
            const int wp0 = w0 + kx - 1;
            const int wp1 = w1 + kx - 1;
            const bool ok0 = hok0 && ((unsigned)wp0 < (unsigned)WW);
            const bool ok1 = hok1 && ((unsigned)wp1 < (unsigned)WW);
            // OOB lanes read from the zero page -> branch-free inner loop
            const unsigned short* xp0 = ok0
                ? X + (size_t)(b0 * HWPIX + hh0 * WW + wp0) * Cin
                : zpad;
            const unsigned short* xp1 = ok1
                ? X + (size_t)(b1 * HWPIX + hh1 * WW + wp1) * Cin
                : zpad;
            const unsigned short* wp =
                Wt + (size_t)((ky * 3 + kx) * Cout) * Cin;
            __builtin_prefetch(xp0, 0, 1);         // global_prefetch_b8
            __builtin_prefetch(xp1, 0, 1);

            for (int c0 = 0; c0 < Cin; c0 += 32) {
                // A fragments: 16x32 bf16, lane needs K runs
                // {half*8 .. +7} and {16+half*8 .. +7}
                Frag16 a0, a1;
                a0.q[0] = *reinterpret_cast<const u32x4*>(xp0 + c0 + koff0);
                a0.q[1] = *reinterpret_cast<const u32x4*>(xp0 + c0 + koff1);
                a1.q[0] = *reinterpret_cast<const u32x4*>(xp1 + c0 + koff0);
                a1.q[1] = *reinterpret_cast<const u32x4*>(xp1 + c0 + koff1);
#pragma unroll
                for (int nt = 0; nt < 4; ++nt) {
                    const int co = coutBase + nt * 16 + nidx;
                    const unsigned short* wq =
                        wp + (size_t)co * Cin + c0 + koff0;
                    Frag16 bf;
                    bf.q[0] = *reinterpret_cast<const u32x4*>(wq);
                    bf.q[1] = *reinterpret_cast<const u32x4*>(wq + 16);
                    acc[0][nt] = __builtin_amdgcn_wmma_f32_16x16x32_bf16(
                        false, a0.v, false, bf.v,
                        (short)0, acc[0][nt], false, false);
                    acc[1][nt] = __builtin_amdgcn_wmma_f32_16x16x32_bf16(
                        false, a1.v, false, bf.v,
                        (short)0, acc[1][nt], false, false);
                }
            }
        }
    }

    // C/D layout: element r of lane L -> M = r + 8*half, N = lane&15
#pragma unroll
    for (int g = 0; g < 2; ++g) {
#pragma unroll
        for (int nt = 0; nt < 4; ++nt) {
            const int n = coutBase + nt * 16 + nidx;
            const float bs = bias[n];
#pragma unroll
            for (int r = 0; r < 8; ++r) {
                const int mm = m0 + g * 16 + r + 8 * half;
                float val = acc[g][nt][r] + bs;
                val = val > 0.f ? val : 0.f;                 // ReLU
                Y[(size_t)mm * Cout + n] = f2bf(val);
            }
        }
    }
}

// ---------------------------------------------------------------------------
// 1x1 head conv (64 -> 54) + box decode. One thread per pixel.
// Weights staged in LDS. Writes boxes [B][N][4] and filtered scores [B][N].
// ---------------------------------------------------------------------------
__global__ __launch_bounds__(256) void head_decode(
    const unsigned short* __restrict__ X3,   // NHWC bf16 [B,H,W,64]
    const float* __restrict__ wo,            // [54][64]
    const float* __restrict__ bo,            // [54]
    float* __restrict__ boxes,               // [B][NBOX][4]
    float* __restrict__ scores)              // [B][NBOX]
{
    __shared__ float sw[54 * 64];
    __shared__ float sb[54];
    for (int i = threadIdx.x; i < 54 * 64; i += 256) sw[i] = wo[i];
    for (int i = threadIdx.x; i < 54; i += 256) sb[i] = bo[i];
    __syncthreads();

    const int pix = blockIdx.x * 256 + threadIdx.x;
    if (pix >= NPIX) return;
    const int b  = pix / HWPIX;
    const int hw = pix - b * HWPIX;
    const int h  = hw / WW;
    const int w  = hw - h * WW;

    float p[54];
#pragma unroll
    for (int o = 0; o < 54; ++o) p[o] = sb[o];
    const unsigned short* xp = X3 + (size_t)pix * 64;
    for (int c = 0; c < 64; ++c) {
        const float xv = bf2f(xp[c]);
#pragma unroll
        for (int o = 0; o < 54; ++o) p[o] = fmaf(xv, sw[o * 64 + c], p[o]);
    }

    const float aw[9] = {0.10101525f, 0.14285714f, 0.20203051f,
                         0.20203051f, 0.28571429f, 0.40406102f,
                         0.40406102f, 0.57142857f, 0.80812204f};
    const float ah[9] = {0.20203051f, 0.14285714f, 0.10101525f,
                         0.40406102f, 0.28571429f, 0.20203051f,
                         0.80812204f, 0.57142857f, 0.40406102f};

#pragma unroll
    for (int a = 0; a < 9; ++a) {
        const float tx = p[a * 6 + 0], ty = p[a * 6 + 1];
        const float tw = p[a * 6 + 2], th = p[a * 6 + 3];
        const float to = p[a * 6 + 4];
        const float sx  = 1.f / (1.f + __expf(-tx));
        const float sy  = 1.f / (1.f + __expf(-ty));
        const float obj = 1.f / (1.f + __expf(-to));
        const float px = (sx + (float)w) * (1.f / (float)WW);
        const float py = (sy + (float)h) * (1.f / (float)HH);
        const float pw = fminf(__expf(tw) * aw[a], 1.f);
        const float ph = fminf(__expf(th) * ah[a], 1.f);
        const size_t bi = (size_t)b * NBOX + (size_t)hw * 9 + a;
        float* ob = boxes + bi * 4;
        ob[0] = px; ob[1] = py; ob[2] = pw; ob[3] = ph;
        scores[bi] = (obj > 0.5f) ? obj : NEGF;   // confidence pre-filter
    }
}

// ---------------------------------------------------------------------------
// Greedy NMS, one workgroup (256 threads) per batch image, 100 iterations.
// outBoxes [B][100][4] f32, outValid [B][100] f32 (1.0 / 0.0).
// ---------------------------------------------------------------------------
__global__ __launch_bounds__(256) void nms_kernel(
    const float* __restrict__ boxes,
    float* __restrict__ scores,              // mutated (rebuilt each launch)
    float* __restrict__ outBoxes,
    float* __restrict__ outValid)
{
    const int b = blockIdx.x;
    const float* bx = boxes + (size_t)b * NBOX * 4;
    float* s = scores + (size_t)b * NBOX;

    __shared__ float sval[256];
    __shared__ int   sidx[256];
    __shared__ float bb[5];                  // x1,y1,x2,y2,area of best
    __shared__ int   bestOk;

    for (int t = 0; t < MAXDET; ++t) {
        // ---- parallel argmax over scores ----
        float mv = -3.0e38f; int mi = 0;
        for (int i = threadIdx.x; i < NBOX; i += 256) {
            const float v = s[i];
            if (v > mv) { mv = v; mi = i; }
        }
        sval[threadIdx.x] = mv; sidx[threadIdx.x] = mi;
        __syncthreads();
        for (int off = 128; off > 0; off >>= 1) {
            if (threadIdx.x < off) {
                if (sval[threadIdx.x + off] > sval[threadIdx.x]) {
                    sval[threadIdx.x] = sval[threadIdx.x + off];
                    sidx[threadIdx.x] = sidx[threadIdx.x + off];
                }
            }
            __syncthreads();
        }
        if (threadIdx.x == 0) {
            const int i = sidx[0];
            const int ok = sval[0] > (NEGF * 0.5f);
            bestOk = ok;
            const float cx = bx[i * 4 + 0], cy = bx[i * 4 + 1];
            const float wd = bx[i * 4 + 2], ht = bx[i * 4 + 3];
            bb[0] = cx - wd * 0.5f; bb[1] = cy - ht * 0.5f;
            bb[2] = cx + wd * 0.5f; bb[3] = cy + ht * 0.5f;
            bb[4] = (bb[2] - bb[0]) * (bb[3] - bb[1]);
            float* ob = outBoxes + ((size_t)b * MAXDET + t) * 4;
            if (ok) {
                ob[0] = cx; ob[1] = cy; ob[2] = wd; ob[3] = ht;
                outValid[b * MAXDET + t] = 1.f;
                s[i] = NEGF;
            } else {
                ob[0] = 0.f; ob[1] = 0.f; ob[2] = 0.f; ob[3] = 0.f;
                outValid[b * MAXDET + t] = 0.f;
            }
        }
        __syncthreads();
        // ---- suppress overlaps ----
        if (bestOk) {
            const float X1 = bb[0], Y1 = bb[1], X2 = bb[2], Y2 = bb[3];
            const float A0 = bb[4];
            for (int i = threadIdx.x; i < NBOX; i += 256) {
                const float cx = bx[i * 4 + 0], cy = bx[i * 4 + 1];
                const float wd = bx[i * 4 + 2], ht = bx[i * 4 + 3];
                const float x1 = cx - wd * 0.5f, y1 = cy - ht * 0.5f;
                const float x2 = cx + wd * 0.5f, y2 = cy + ht * 0.5f;
                const float xx1 = fmaxf(X1, x1), yy1 = fmaxf(Y1, y1);
                const float xx2 = fminf(X2, x2), yy2 = fminf(Y2, y2);
                const float inter = fmaxf(xx2 - xx1, 0.f) * fmaxf(yy2 - yy1, 0.f);
                const float area = (x2 - x1) * (y2 - y1);
                const float iou = inter / (A0 + area - inter + 1e-16f);
                if (iou > 0.5f) s[i] = NEGF;
            }
        }
        __syncthreads();
    }
}

// ---------------------------------------------------------------------------
// Host-side launcher
// ---------------------------------------------------------------------------
extern "C" void kernel_launch(void* const* d_in, const int* in_sizes, int n_in,
                              void* d_out, int out_size, void* d_ws, size_t ws_size,
                              hipStream_t stream)
{
    (void)in_sizes; (void)n_in; (void)out_size; (void)ws_size;

    const float* features = (const float*)d_in[0];
    const float* w1 = (const float*)d_in[1];
    const float* b1 = (const float*)d_in[2];
    const float* g1 = (const float*)d_in[3];
    const float* be1 = (const float*)d_in[4];
    const float* m1 = (const float*)d_in[5];
    const float* v1 = (const float*)d_in[6];
    const float* w2 = (const float*)d_in[7];
    const float* b2 = (const float*)d_in[8];
    const float* g2 = (const float*)d_in[9];
    const float* be2 = (const float*)d_in[10];
    const float* m2 = (const float*)d_in[11];
    const float* v2 = (const float*)d_in[12];
    const float* w3 = (const float*)d_in[13];
    const float* b3 = (const float*)d_in[14];
    const float* g3 = (const float*)d_in[15];
    const float* be3 = (const float*)d_in[16];
    const float* m3 = (const float*)d_in[17];
    const float* v3 = (const float*)d_in[18];
    const float* wo = (const float*)d_in[19];
    const float* bo = (const float*)d_in[20];

    // ---- workspace carve-out ----
    char* ws = (char*)d_ws;
    size_t off = 0;
    auto carve = [&](size_t bytes) -> char* {
        char* p = ws + off;
        off = (off + bytes + 255) & ~(size_t)255;
        return p;
    };
    unsigned short* featB = (unsigned short*)carve((size_t)NPIX * 512 * 2);
    unsigned short* x1    = (unsigned short*)carve((size_t)NPIX * 256 * 2);
    unsigned short* x2    = (unsigned short*)carve((size_t)NPIX * 128 * 2);
    unsigned short* x3    = (unsigned short*)carve((size_t)NPIX * 64 * 2);
    unsigned short* w1f   = (unsigned short*)carve((size_t)9 * 256 * 512 * 2);
    unsigned short* w2f   = (unsigned short*)carve((size_t)9 * 128 * 256 * 2);
    unsigned short* w3f   = (unsigned short*)carve((size_t)9 * 64 * 128 * 2);
    float* b1f = (float*)carve(256 * 4);
    float* b2f = (float*)carve(128 * 4);
    float* b3f = (float*)carve(64 * 4);
    unsigned short* zpad  = (unsigned short*)carve(PADELEMS * 2);
    float* boxes  = (float*)carve((size_t)BD * NBOX * 4 * 4);
    float* scores = (float*)carve((size_t)BD * NBOX * 4);

    // ---- 0) zero the OOB pad page ----
    zero_pad<<<1, 256, 0, stream>>>(zpad);

    // ---- 1) weight folding (BN fused) + repack ----
    {
        int tot = 256 * 512 * 9;
        fold_weights<<<(tot + 255) / 256, 256, 0, stream>>>(
            w1, b1, g1, be1, m1, v1, w1f, b1f, 512, 256);
        tot = 128 * 256 * 9;
        fold_weights<<<(tot + 255) / 256, 256, 0, stream>>>(
            w2, b2, g2, be2, m2, v2, w2f, b2f, 256, 128);
        tot = 64 * 128 * 9;
        fold_weights<<<(tot + 255) / 256, 256, 0, stream>>>(
            w3, b3, g3, be3, m3, v3, w3f, b3f, 128, 64);
    }

    // ---- 2) features NCHW f32 -> NHWC bf16 ----
    {
        size_t tot = (size_t)NPIX * 512;
        nchw_to_nhwc_bf16<<<(unsigned)((tot + 255) / 256), 256, 0, stream>>>(
            features, featB, 512);
    }

    // ---- 3) conv stack (implicit GEMM, WMMA bf16) ----
    // waves = Mblocks(32-row) * (Cout/64); 8 waves per block
    const int mblocks = NPIX / 32;                  // 1568
    conv3x3_wmma<<<mblocks * (256 / 64) / 8, 256, 0, stream>>>(
        featB, w1f, b1f, zpad, x1, 512, 256);
    conv3x3_wmma<<<mblocks * (128 / 64) / 8, 256, 0, stream>>>(
        x1, w2f, b2f, zpad, x2, 256, 128);
    conv3x3_wmma<<<mblocks * (64 / 64) / 8, 256, 0, stream>>>(
        x2, w3f, b3f, zpad, x3, 128, 64);

    // ---- 4) 1x1 head + decode ----
    head_decode<<<(NPIX + 255) / 256, 256, 0, stream>>>(
        x3, wo, bo, boxes, scores);

    // ---- 5) per-image greedy NMS ----
    float* outBoxes = (float*)d_out;                       // [16][100][4]
    float* outValid = (float*)d_out + (size_t)BD * MAXDET * 4;   // [16][100]
    nms_kernel<<<BD, 256, 0, stream>>>(boxes, scores, outBoxes, outValid);
}
